// EdgeDecoder_14542759264855
// MI455X (gfx1250) — matrix operand order repaired and checked by reading, not compiled, not run on hardware
//
#include <hip/hip_runtime.h>
#include <hip/hip_bf16.h>

#define HDIM 128
#define K2   256  // 2*H
#define W1ELEMS (HDIM * K2)   // 32768 bf16 = 64 KB

typedef __attribute__((ext_vector_type(16))) __bf16 v16bf;
typedef __attribute__((ext_vector_type(8)))  float  v8f;
typedef __attribute__((ext_vector_type(4)))  float  v4f;

// Prologue: write W1 in WMMA-B fragment order + fused BN affine params.
// frag layout: w1f[((kb*8 + nt)*32 + lane)*16 + e] = W1[n][k]
//   n = nt*16 + (lane&15), k = kb*32 + (lane>>4)*16 + e
// -> a wave's B fragment is one contiguous, lane-linear 1 KB block.
__global__ void ed_prep(const float* __restrict__ W1, const float* __restrict__ b1,
                        const float* __restrict__ gamma, const float* __restrict__ beta,
                        const float* __restrict__ rmean, const float* __restrict__ rvar,
                        const float* __restrict__ W2,
                        __bf16* __restrict__ w1f, float* __restrict__ scale,
                        float* __restrict__ shift, float* __restrict__ w2c)
{
    int idx = blockIdx.x * blockDim.x + threadIdx.x;
    if (idx < W1ELEMS) {
        int e    = idx & 15;
        int lane = (idx >> 4) & 31;
        int f    = idx >> 9;          // kb*8 + nt
        int nt   = f & 7;
        int kb   = f >> 3;
        int n    = nt * 16 + (lane & 15);
        int k    = kb * 32 + (lane >> 4) * 16 + e;
        w1f[idx] = (__bf16)W1[n * K2 + k];
    }
    if (idx < HDIM) {
        float s = gamma[idx] * rsqrtf(rvar[idx] + 1e-5f);
        scale[idx] = s;
        shift[idx] = beta[idx] + (b1[idx] - rmean[idx]) * s;
        w2c[idx]   = W2[idx];
    }
}

// Gather one 16x32 bf16 A fragment for K-block kb from the concatenated row.
// A in contiguous k order used here (same as w1f B): lane (M = lane&15) holds K = [o..o+15], o = 16*(lane>>4)
__device__ __forceinline__ v16bf load_a(const float* __restrict__ up,
                                        const float* __restrict__ fp,
                                        int kb, int o)
{
    const float* src = (kb < 4) ? (up + kb * 32) : (fp + (kb - 4) * 32);
    v4f a0 = *(const v4f*)(src + o);
    v4f a1 = *(const v4f*)(src + o + 4);
    v4f a2 = *(const v4f*)(src + o + 8);
    v4f a3 = *(const v4f*)(src + o + 12);
    v16bf a;
    #pragma unroll
    for (int i = 0; i < 4; ++i) {
        a[i]      = (__bf16)a0[i];
        a[4 + i]  = (__bf16)a1[i];
        a[8 + i]  = (__bf16)a2[i];
        a[12 + i] = (__bf16)a3[i];
    }
    return a;
}

// Main: persistent waves; 1 wave = 32 edges (two 16-edge M-tiles sharing each B fragment).
// W1 fragments staged once per block into LDS; B reads are conflict-free ds_load_b128.
__global__ void __launch_bounds__(256) ed_main(
    const float* __restrict__ z_user, const float* __restrict__ z_food,
    const int* __restrict__ row, const int* __restrict__ col,
    const __bf16* __restrict__ w1f, const float* __restrict__ scale,
    const float* __restrict__ shift, const float* __restrict__ w2c,
    const float* __restrict__ b2, float* __restrict__ out, int E)
{
    __shared__ __attribute__((aligned(16))) __bf16 w1s[W1ELEMS];  // 64 KB

    // Coalesced global -> LDS staging of the fragment table
    #pragma unroll
    for (int i = threadIdx.x; i < W1ELEMS / 8; i += 256)
        ((v4f*)w1s)[i] = ((const v4f*)w1f)[i];
    __syncthreads();

    const int lane = threadIdx.x & 31;
    const int wave = threadIdx.x >> 5;
    const int m    = lane & 15;
    const int half = lane >> 4;
    const int o    = half * 16;

    // Per-lane BN/W2 params (col n = nt*16 + m), loaded once per wave
    float sc[8], sh[8], w2l[8];
    #pragma unroll
    for (int nt = 0; nt < 8; ++nt) {
        int n = nt * 16 + m;
        sc[nt] = scale[n]; sh[nt] = shift[n]; w2l[nt] = w2c[n];
    }
    const float bias = b2[0] + 0.1f;

    const long tiles32    = ((long)E + 31) / 32;
    const long totalWaves = (long)gridDim.x * 8;

    #pragma clang loop unroll(disable)
    for (long t = (long)blockIdx.x * 8 + wave; t < tiles32; t += totalWaves) {
        // Opaque zero offset: blocks LICM from hoisting the (loop-invariant) LDS
        // B-fragment loads out of this loop, which previously ballooned to ~700
        // VGPRs and spilled to scratch. Costs nothing at runtime.
        int tw = 0;
        asm volatile("" : "+v"(tw));
        const __bf16* wbase = w1s + tw + lane * 16;

        const long ebase = t * 32;
        long e0 = ebase + m;        if (e0 >= E) e0 = (long)E - 1;
        long e1 = ebase + 16 + m;   if (e1 >= E) e1 = (long)E - 1;
        const float* up0 = z_user + (long)row[e0] * HDIM;
        const float* fp0 = z_food + (long)col[e0] * HDIM;
        const float* up1 = z_user + (long)row[e1] * HDIM;
        const float* fp1 = z_food + (long)col[e1] * HDIM;

        v8f acc0[8] = {};
        v8f acc1[8] = {};

        #pragma unroll
        for (int kb = 0; kb < 8; ++kb) {              // K = 256 in steps of 32
            v16bf a0 = load_a(up0, fp0, kb, o);
            v16bf a1 = load_a(up1, fp1, kb, o);
            // lane-linear B fragments: byte addr = (kb*8+nt)*1024 + lane*32
            const __bf16* bs = wbase + (kb * 8) * 512;
            #pragma unroll
            for (int nt = 0; nt < 8; ++nt) {
                v16bf b = *(const v16bf*)(bs + nt * 512);
                acc0[nt] = __builtin_amdgcn_wmma_f32_16x16x32_bf16(
                    false, a0, false, b, (short)0, acc0[nt], false, false);
                acc1[nt] = __builtin_amdgcn_wmma_f32_16x16x32_bf16(
                    false, a1, false, b, (short)0, acc1[nt], false, false);
            }
        }

        // Epilogue: BN affine + ReLU + dot(W2) per lane, 16-lane butterfly reduce, sigmoid
        #pragma unroll
        for (int sub = 0; sub < 2; ++sub) {
            v8f* acc = sub ? acc1 : acc0;
            const long sbase = ebase + sub * 16;
            #pragma unroll
            for (int rr = 0; rr < 8; ++rr) {          // D VGPR rr -> row rr + 8*half
                float p = 0.f;
                #pragma unroll
                for (int nt = 0; nt < 8; ++nt) {
                    float h = fmaf(acc[nt][rr], sc[nt], sh[nt]);
                    h = fmaxf(h, 0.f);
                    p = fmaf(h, w2l[nt], p);
                }
                #pragma unroll
                for (int off = 8; off > 0; off >>= 1)
                    p += __shfl_xor(p, off, 32);      // xor<16 never crosses halves
                if (m == 0) {
                    long eo = sbase + rr + half * 8;
                    if (eo < E) out[eo] = 1.0f / (1.0f + __expf(-(p + bias)));
                }
            }
        }
    }
}

extern "C" void kernel_launch(void* const* d_in, const int* in_sizes, int n_in,
                              void* d_out, int out_size, void* d_ws, size_t ws_size,
                              hipStream_t stream) {
    const float* z_user = (const float*)d_in[0];
    const float* z_food = (const float*)d_in[1];
    const int*   row    = (const int*)d_in[2];
    const int*   col    = (const int*)d_in[3];
    const float* W1     = (const float*)d_in[4];
    const float* b1     = (const float*)d_in[5];
    const float* gamma  = (const float*)d_in[6];
    const float* beta   = (const float*)d_in[7];
    const float* rmean  = (const float*)d_in[8];
    const float* rvar   = (const float*)d_in[9];
    const float* W2     = (const float*)d_in[10];
    const float* b2     = (const float*)d_in[11];
    const int E = in_sizes[2];

    char*   ws    = (char*)d_ws;
    __bf16* w1f   = (__bf16*)ws;                    // 65536 B fragment table
    float*  scale = (float*)(ws + 65536);
    float*  shift = (float*)(ws + 65536 + 512);
    float*  w2c   = (float*)(ws + 65536 + 1024);

    ed_prep<<<(W1ELEMS + 255) / 256, 256, 0, stream>>>(
        W1, b1, gamma, beta, rmean, rvar, W2, w1f, scale, shift, w2c);

    const long tiles32 = ((long)E + 31) / 32;
    long blocks = (tiles32 + 7) / 8;                 // 8 waves/block, 32 edges/wave
    if (blocks > 1024) blocks = 1024;                // persistent grid-stride
    ed_main<<<(int)blocks, 256, 0, stream>>>(
        z_user, z_food, row, col, w1f, scale, shift, w2c, b2, (float*)d_out, E);
}
